// MambaLayer_1812476199142
// MI455X (gfx1250) — compile-verified, matrix operand order checked by reading, not run on hardware
//
#include <hip/hip_runtime.h>

#define BB   8
#define CDIM 128
#define HH   64
#define WW   64
#define NN   4096          // H*W
#define DI   256           // D_INNER
#define DS   16            // D_STATE
#define DTR  8             // DT_RANK
#define ROWS (BB*NN)       // 32768
#define XPW  48            // x_proj out cols padded 40 -> 48
#define NCH  32            // scan chunks
#define CLEN 128           // chunk length (NN / NCH)

typedef __bf16 bf16;
typedef __bf16 v16bf __attribute__((ext_vector_type(16)));
typedef float  v8f   __attribute__((ext_vector_type(8)));

__device__ __forceinline__ bf16 f2bf(float f) {
  unsigned u = __builtin_bit_cast(unsigned, f);
  u += 0x7fffu + ((u >> 16) & 1u);                 // round-to-nearest-even
  unsigned short s = (unsigned short)(u >> 16);
  return __builtin_bit_cast(bf16, s);
}
__device__ __forceinline__ float bf2f(bf16 h) {
  unsigned u = ((unsigned)__builtin_bit_cast(unsigned short, h)) << 16;
  return __builtin_bit_cast(float, u);
}
__device__ __forceinline__ float siluf(float x) { return x / (1.0f + __expf(-x)); }

// ---------------------------------------------------------------- converts
__global__ void k_f32bf(const float* __restrict__ s, bf16* __restrict__ d, int n) {
  int i = blockIdx.x * 256 + threadIdx.x;
  if (i < n) d[i] = f2bf(s[i]);
}

__global__ void k_xproj_pad(const float* __restrict__ s, bf16* __restrict__ d) {
  int i = blockIdx.x * 256 + threadIdx.x;          // [XPW][DI] row-major
  if (i >= XPW * DI) return;
  int r = i >> 8;
  d[i] = (r < DTR + 2 * DS) ? f2bf(s[i]) : f2bf(0.0f);
}

// -------------------------------------------- snake gather + PE + layernorm
// one wave per (b,n) row; writes bf16 [ROWS][CDIM]
__global__ void k_prep_ln(const float* __restrict__ x, const float* __restrict__ pe,
                          const float* __restrict__ g, const float* __restrict__ be,
                          bf16* __restrict__ out, int dir) {
  int row  = (blockIdx.x * 256 + threadIdx.x) >> 5;   // 0..ROWS-1
  int lane = threadIdx.x & 31;
  if (row >= ROWS) return;
  int b = row >> 12;
  int n = row & (NN - 1);
  int h, w;
  if (dir == 0) { h = n >> 6; int wp = n & 63; w = (h & 1) ? (63 - wp) : wp; }
  else          { w = n >> 6; int hp = n & 63; h = (w & 1) ? (63 - hp) : hp; }
  const float* xb = x + (size_t)b * (CDIM * HH * WW) + h * WW + w;  // + c*4096
  float v[4]; float s = 0.0f;
  #pragma unroll
  for (int i = 0; i < 4; ++i) {
    int c = lane + 32 * i;
    v[i] = xb[(size_t)c * (HH * WW)] + pe[(size_t)n * CDIM + c];
    s += v[i];
  }
  for (int off = 16; off >= 1; off >>= 1) s += __shfl_xor(s, off, 32);
  float mean = s * (1.0f / CDIM);
  float vs = 0.0f;
  #pragma unroll
  for (int i = 0; i < 4; ++i) { float d = v[i] - mean; vs += d * d; }
  for (int off = 16; off >= 1; off >>= 1) vs += __shfl_xor(vs, off, 32);
  float inv = rsqrtf(vs * (1.0f / CDIM) + 1e-5f);
  #pragma unroll
  for (int i = 0; i < 4; ++i) {
    int c = lane + 32 * i;
    out[(size_t)row * CDIM + c] = f2bf((v[i] - mean) * inv * g[c] + be[c]);
  }
}

// ------------------------------------------------------- WMMA bf16 GEMM
// C[M,N] = A[M,K] @ W[N,K]^T ; K % 32 == 0, K <= 256.
// Block = 256 threads (8 waves). Each block: one 16-row M tile, 8 N tiles.
// The 16xK A-tile (contiguous in memory) is staged into LDS once with
// CDNA5 async global->LDS loads, then shared by all 8 waves.
__global__ void __launch_bounds__(256)
k_gemm_bf16(const bf16* __restrict__ A, const bf16* __restrict__ W,
            float* __restrict__ C, int M, int N, int K) {
  __shared__ __align__(16) bf16 smA[16 * 256];     // up to 8 KB used
  int tid  = threadIdx.x;
  int wave = tid >> 5;
  int lane = tid & 31;
  int mT   = blockIdx.x;
  int nTi  = blockIdx.y * 8 + wave;

  // ---- async stage A tile (16*K*2 bytes, contiguous) into LDS ----
  unsigned bytes = (unsigned)(16 * K * 2);
  unsigned long long gbase =
      (unsigned long long)(const char*)(A + (size_t)mT * 16 * K);
  unsigned lbase = (unsigned)(size_t)(&smA[0]);
  for (unsigned ofs = (unsigned)tid * 16u; ofs < bytes; ofs += 256u * 16u) {
    unsigned laddr = lbase + ofs;
    unsigned long long gaddr = gbase + ofs;
    asm volatile("global_load_async_to_lds_b128 %0, %1, off"
                 :: "v"(laddr), "v"(gaddr) : "memory");
  }
  asm volatile("s_wait_asynccnt 0x0" ::: "memory");
  __syncthreads();

  if (nTi * 16 < N) {
    int half = lane >> 4, l16 = lane & 15;
    const bf16* ar = smA + (size_t)l16 * K;             // LDS reads
    const bf16* br = W + (size_t)(nTi * 16 + l16) * K;  // weights via L2
    v8f acc = {};
    for (int k0 = 0; k0 < K; k0 += 32) {
      int ka = k0 + half * 8;  // lanes 0-15: K 0..7/16..23; lanes 16-31: +8
      v16bf a, bm;
      #pragma unroll
      for (int i = 0; i < 8; ++i) {
        a[i]      = ar[ka + i];
        a[i + 8]  = ar[ka + 16 + i];
        bm[i]     = br[ka + i];
        bm[i + 8] = br[ka + 16 + i];
      }
      acc = __builtin_amdgcn_wmma_f32_16x16x32_bf16(false, a, false, bm,
                                                    (short)0, acc, false, false);
    }
    // D layout: vgpr v, lanes 0-15 -> (m=v,n=l16); lanes 16-31 -> (m=v+8,n=l16)
    float* cr = C + (size_t)(mT * 16 + half * 8) * N + nTi * 16 + l16;
    #pragma unroll
    for (int v = 0; v < 8; ++v) cr[(size_t)v * N] = acc[v];
  }
}

// ----------------------------------------------- depthwise conv(4) + SiLU
__global__ void k_conv_silu(const float* __restrict__ xz, const float* __restrict__ cw,
                            const float* __restrict__ cb, bf16* __restrict__ u) {
  size_t i = (size_t)blockIdx.x * 256 + threadIdx.x;    // over ROWS*DI
  if (i >= (size_t)ROWS * DI) return;
  int d = (int)(i & (DI - 1));
  size_t row = i >> 8;
  int l = (int)(row & (NN - 1));
  size_t rowb = row - l;
  float acc = cb[d];
  #pragma unroll
  for (int k = 0; k < 4; ++k) {
    int j = l + k - 3;
    if (j >= 0) acc += xz[(rowb + j) * (size_t)(2 * DI) + d] * cw[d * 4 + k];
  }
  u[i] = f2bf(siluf(acc));
}

// ------------------------------------------------------- dt = softplus(...)
__global__ void k_dt(const float* __restrict__ xdbl, const float* __restrict__ dtw,
                     const float* __restrict__ dtb, float* __restrict__ dt) {
  size_t i = (size_t)blockIdx.x * 256 + threadIdx.x;
  if (i >= (size_t)ROWS * DI) return;
  int d = (int)(i & (DI - 1));
  size_t row = i >> 8;
  const float* xr = xdbl + row * XPW;
  float acc = dtb[d];
  #pragma unroll
  for (int r = 0; r < DTR; ++r) acc += xr[r] * dtw[d * DTR + r];
  dt[i] = (acc > 20.0f) ? acc : log1pf(__expf(acc));
}

// ---------------------------------------- chunked associative selective scan
// thread id g = (((b*DI+d)*NCH+ch)*DS+s) ; 16 consecutive lanes = states s
__global__ void k_scan_chunk(const float* __restrict__ dt, const bf16* __restrict__ u,
                             const float* __restrict__ xdbl, const float* __restrict__ A_log,
                             float* __restrict__ cA, float* __restrict__ cQ) {
  int g = blockIdx.x * 256 + threadIdx.x;
  int s  = g & 15;
  int ch = (g >> 4) & 31;
  int d  = (g >> 9) & 255;
  int b  = g >> 17;
  float Av = -__expf(A_log[d * DS + s]);
  float a = 1.0f, q = 0.0f;
  size_t row0 = (size_t)b * NN + (size_t)ch * CLEN;
  for (int t = 0; t < CLEN; ++t) {
    size_t row = row0 + t;
    const float* xr = xdbl + row * XPW;
    __builtin_prefetch(xr + XPW, 0, 0);
    float dtv = dt[row * DI + d];
    float uv  = bf2f(u[row * DI + d]);
    float dA  = __expf(dtv * Av);
    q = dA * q + (dtv * uv) * xr[DTR + s];
    a *= dA;
  }
  int idx = ((b * DI + d) * DS + s) * NCH + ch;
  cA[idx] = a; cQ[idx] = q;
}

__global__ void k_scan_combine(const float* __restrict__ cA, const float* __restrict__ cQ,
                               float* __restrict__ cH) {
  int g = blockIdx.x * 256 + threadIdx.x;    // (b*DI+d)*DS+s : 32768
  if (g >= BB * DI * DS) return;
  const float* a = cA + (size_t)g * NCH;
  const float* q = cQ + (size_t)g * NCH;
  float* hh = cH + (size_t)g * NCH;
  float h = 0.0f;
  for (int c = 0; c < NCH; ++c) { hh[c] = h; h = a[c] * h + q[c]; }
}

__global__ void k_scan_emit(const float* __restrict__ dt, const bf16* __restrict__ u,
                            const float* __restrict__ xdbl, const float* __restrict__ A_log,
                            const float* __restrict__ cH, float* __restrict__ ys) {
  int g = blockIdx.x * 256 + threadIdx.x;
  int s  = g & 15;
  int ch = (g >> 4) & 31;
  int d  = (g >> 9) & 255;
  int b  = g >> 17;
  float Av = -__expf(A_log[d * DS + s]);
  float h = cH[((b * DI + d) * DS + s) * NCH + ch];
  size_t row0 = (size_t)b * NN + (size_t)ch * CLEN;
  for (int t = 0; t < CLEN; ++t) {
    size_t row = row0 + t;
    const float* xr = xdbl + row * XPW;
    __builtin_prefetch(xr + XPW, 0, 0);
    float dtv = dt[row * DI + d];
    float uv  = bf2f(u[row * DI + d]);
    float dA  = __expf(dtv * Av);
    h = dA * h + (dtv * uv) * xr[DTR + s];
    float y = h * xr[DTR + DS + s];
    y += __shfl_xor(y, 1, 32);
    y += __shfl_xor(y, 2, 32);
    y += __shfl_xor(y, 4, 32);
    y += __shfl_xor(y, 8, 32);
    if (s == 0) ys[row * DI + d] = y;
  }
}

// ------------------------------------------------------------------ gating
__global__ void k_gate(const float* __restrict__ ys, const bf16* __restrict__ u,
                       const float* __restrict__ Dp, const float* __restrict__ xz,
                       bf16* __restrict__ yg) {
  size_t i = (size_t)blockIdx.x * 256 + threadIdx.x;
  if (i >= (size_t)ROWS * DI) return;
  int d = (int)(i & (DI - 1));
  size_t row = i >> 8;
  float z = xz[row * (size_t)(2 * DI) + DI + d];
  float y = ys[i] + bf2f(u[i]) * Dp[d];
  yg[i] = f2bf(y * siluf(z));
}

// ----------------------------------------------- un-snake both + sum -> out
__global__ void k_combine(const float* __restrict__ y1, const float* __restrict__ y2,
                          float* __restrict__ out) {
  size_t i = (size_t)blockIdx.x * 256 + threadIdx.x;   // (B,C,H,W)
  if (i >= (size_t)BB * CDIM * HH * WW) return;
  int w = (int)(i & 63);
  int h = (int)((i >> 6) & 63);
  int c = (int)((i >> 12) & 127);
  int b = (int)(i >> 19);
  int n1 = h * WW + ((h & 1) ? (63 - w) : w);
  int n2 = w * HH + ((w & 1) ? (63 - h) : h);
  size_t r1 = ((size_t)b * NN + n1) * CDIM + c;
  size_t r2 = ((size_t)b * NN + n2) * CDIM + c;
  out[i] = y1[r1] + y2[r2];
}

// ====================================================================== host
extern "C" void kernel_launch(void* const* d_in, const int* in_sizes, int n_in,
                              void* d_out, int out_size, void* d_ws, size_t ws_size,
                              hipStream_t stream) {
  (void)in_sizes; (void)n_in; (void)out_size; (void)ws_size;
  const float* x  = (const float*)d_in[0];
  const float* pe = (const float*)d_in[1];
  const float* lng[2] = { (const float*)d_in[3], (const float*)d_in[5] };
  const float* lnb[2] = { (const float*)d_in[4], (const float*)d_in[6] };

  // workspace carve-up (256B aligned)
  char* p = (char*)d_ws;
  auto alloc = [&](size_t bytes) -> void* {
    void* r = (void*)p;
    p += (bytes + 255) & ~(size_t)255;
    return r;
  };
  bf16*  xln   = (bf16*) alloc((size_t)ROWS * CDIM * 2);        // 8 MB
  float* xz    = (float*)alloc((size_t)ROWS * 2 * DI * 4);      // 64 MB
  bf16*  ubf   = (bf16*) alloc((size_t)ROWS * DI * 2);          // 16 MB
  float* xdbl  = (float*)alloc((size_t)ROWS * XPW * 4);         // 6 MB
  float* dtbuf = (float*)alloc((size_t)ROWS * DI * 4);          // 32 MB
  float* ys    = (float*)alloc((size_t)ROWS * DI * 4);          // 32 MB
  bf16*  ygbf  = (bf16*) alloc((size_t)ROWS * DI * 2);          // 16 MB
  float* yout1 = (float*)alloc((size_t)ROWS * CDIM * 4);        // 16 MB
  float* yout2 = (float*)alloc((size_t)ROWS * CDIM * 4);        // 16 MB
  float* cA    = (float*)alloc((size_t)BB * DI * DS * NCH * 4); // 4 MB
  float* cQ    = (float*)alloc((size_t)BB * DI * DS * NCH * 4);
  float* cH    = (float*)alloc((size_t)BB * DI * DS * NCH * 4);
  bf16*  inwb  = (bf16*) alloc((size_t)2 * DI * CDIM * 2);
  bf16*  xpwb  = (bf16*) alloc((size_t)XPW * DI * 2);
  bf16*  outwb = (bf16*) alloc((size_t)CDIM * DI * 2);

  for (int br = 0; br < 2; ++br) {
    int base = 7 + br * 9;
    const float* in_w    = (const float*)d_in[base + 0];
    const float* conv_w  = (const float*)d_in[base + 1];
    const float* conv_b  = (const float*)d_in[base + 2];
    const float* xproj_w = (const float*)d_in[base + 3];
    const float* dt_w    = (const float*)d_in[base + 4];
    const float* dt_b    = (const float*)d_in[base + 5];
    const float* A_log   = (const float*)d_in[base + 6];
    const float* Dp      = (const float*)d_in[base + 7];
    const float* out_w   = (const float*)d_in[base + 8];
    float* yout = br ? yout2 : yout1;

    // weight conversion (bf16 for WMMA)
    k_f32bf    <<<(2 * DI * CDIM + 255) / 256, 256, 0, stream>>>(in_w, inwb, 2 * DI * CDIM);
    k_xproj_pad<<<(XPW * DI + 255) / 256,      256, 0, stream>>>(xproj_w, xpwb);
    k_f32bf    <<<(CDIM * DI + 255) / 256,     256, 0, stream>>>(out_w, outwb, CDIM * DI);

    // 1. snake gather + PE + layernorm -> bf16
    k_prep_ln<<<ROWS / 8, 256, 0, stream>>>(x, pe, lng[br], lnb[br], xln, br);
    // 2. in_proj: [32768,128] x [512,128]^T -> xz f32
    k_gemm_bf16<<<dim3(ROWS / 16, (2 * DI + 127) / 128), 256, 0, stream>>>(
        xln, inwb, xz, ROWS, 2 * DI, CDIM);
    // 3. depthwise conv + silu -> u bf16
    k_conv_silu<<<(ROWS * DI) / 256, 256, 0, stream>>>(xz, conv_w, conv_b, ubf);
    // 4. x_proj: [32768,256] x [48,256]^T -> x_dbl f32
    k_gemm_bf16<<<dim3(ROWS / 16, (XPW + 127) / 128), 256, 0, stream>>>(
        ubf, xpwb, xdbl, ROWS, XPW, DI);
    // 5. dt = softplus(x_dbl[:, :8] @ dt_w^T + dt_b)
    k_dt<<<(ROWS * DI) / 256, 256, 0, stream>>>(xdbl, dt_w, dt_b, dtbuf);
    // 6. chunked selective scan
    k_scan_chunk  <<<(BB * DI * DS * NCH) / 256, 256, 0, stream>>>(dtbuf, ubf, xdbl, A_log, cA, cQ);
    k_scan_combine<<<(BB * DI * DS) / 256,       256, 0, stream>>>(cA, cQ, cH);
    k_scan_emit   <<<(BB * DI * DS * NCH) / 256, 256, 0, stream>>>(dtbuf, ubf, xdbl, A_log, cH, ys);
    // 7. gate: (ys + u*Dp) * silu(z) -> bf16
    k_gate<<<(ROWS * DI) / 256, 256, 0, stream>>>(ys, ubf, Dp, xz, ygbf);
    // 8. out_proj: [32768,256] x [128,256]^T -> yout f32
    k_gemm_bf16<<<dim3(ROWS / 16, (CDIM + 127) / 128), 256, 0, stream>>>(
        ygbf, outwb, yout, ROWS, CDIM, DI);
  }

  // un-snake both branches and sum into d_out
  k_combine<<<(BB * CDIM * HH * WW) / 256, 256, 0, stream>>>(yout1, yout2, (float*)d_out);
}
